// ResonanceEmbedder_35253091566219
// MI455X (gfx1250) — matrix operand order repaired and checked by reading, not compiled, not run on hardware
//
#include <hip/hip_runtime.h>
#include <math.h>

// ---------------- problem constants ----------------
constexpr int D     = 256;
constexpr int SEQ   = 4096;
constexpr int BATCH = 8;
constexpr int NROWS = BATCH * SEQ;   // 32768
constexpr int CW    = 5;

// LDS tile row strides (elements). 40 elems = 80 B = 20 words:
// 16 consecutive columns start 20 words apart -> all 64 banks covered, conflict-free b128 reads.
// The TDM pad feature (64B data + 16B pad per row) produces exactly this stride.
constexpr int LDS_BSTR = 40;
constexpr int LDS_ASTR = 40;

// ---------------- CDNA5 WMMA types ----------------
typedef __attribute__((ext_vector_type(16))) __bf16 v16bf;
typedef __attribute__((ext_vector_type(8)))  __bf16 v8bf;
typedef __attribute__((ext_vector_type(8)))  float  v8f;

union Frag {
  v16bf v;
  v8bf  h[2];
  unsigned long long q[4];
};

union Half8 {
  v8bf v;
  unsigned long long q[2];
};

__device__ __forceinline__ unsigned short f32_to_bf16(float f) {
  unsigned int u = __float_as_uint(f);
  u += 0x7fffu + ((u >> 16) & 1u);       // round-to-nearest-even
  return (unsigned short)(u >> 16);
}

// ---------------- Tensor Data Mover / async fallback ----------------
#if defined(__has_builtin)
#if __has_builtin(__builtin_amdgcn_tensor_load_to_lds)
#define TDM_AVAILABLE 1
#endif
#endif
#ifndef TDM_AVAILABLE
#define TDM_AVAILABLE 0
#endif

typedef __attribute__((ext_vector_type(4))) unsigned tdm_v4u;
typedef __attribute__((ext_vector_type(8))) int      tdm_v8i;
typedef __attribute__((ext_vector_type(4))) int      tdm_v4i;

// One TDM descriptor: load a 256(row) x 32(K) bf16 tile from an N-major weight
// matrix (row stride = kdim elements) into LDS, padded 64B data + 16B pad per row.
__device__ __forceinline__ void tdm_load_b_tile(unsigned lds_addr,
                                                const unsigned short* gsrc, int kdim) {
#if TDM_AVAILABLE
  unsigned long long ga = (unsigned long long)(size_t)gsrc;
  tdm_v4u g0;
  g0[0] = 1u;                                                 // count=1 (valid user D#)
  g0[1] = lds_addr;                                           // LDS byte address
  g0[2] = (unsigned)ga;                                       // global_addr[31:0]
  g0[3] = ((unsigned)(ga >> 32) & 0x01FFFFFFu) | 0x80000000u; // addr[56:32] | type=2

  union { tdm_v8i v; unsigned u[8]; } g1;
  g1.u[0] = (1u << 16)                       // data_size = 2 bytes
          | (1u << 20)                       // pad_enable
          | (3u << 22)                       // pad_interval code 3 -> every 16 DWORDs (64 B)
          | (3u << 25);                      // pad_amount  code 3 -> 4 DWORDs (16 B)
  g1.u[1] = ((unsigned)kdim & 0xFFFFu) << 16;          // tensor_dim0[15:0]
  g1.u[2] = (((unsigned)kdim >> 16) & 0xFFFFu)         // tensor_dim0[31:16]
          | (256u << 16);                              // tensor_dim1[15:0]
  g1.u[3] = (32u << 16);                               // tensor_dim1[31:16]=0 | tile_dim0=32
  g1.u[4] = 256u;                                      // tile_dim1=256, tile_dim2=0
  g1.u[5] = (unsigned)kdim;                            // tensor_dim0_stride[31:0]
  g1.u[6] = 0u;                                        // stride[47:32] | dim1_stride[15:0]
  g1.u[7] = 0u;

  tdm_v4i z4 = {};
#if __has_include(<hip/amd_detail/amd_gfx1250_TDM.h>)
  tdm_v8i z8 = {};
  __builtin_amdgcn_tensor_load_to_lds(g0, g1.v, z4, z4, z8, 0);
#else
  __builtin_amdgcn_tensor_load_to_lds(g0, g1.v, z4, z4, 0);
#endif
#else
  (void)lds_addr; (void)gsrc; (void)kdim;
#endif
}

// async 16-byte global -> LDS copy (fallback path)
__device__ __forceinline__ void async_copy16(unsigned lds_off, unsigned long long gaddr) {
  asm volatile("global_load_async_to_lds_b128 %0, %1, off"
               :: "v"(lds_off), "v"(gaddr) : "memory");
}
__device__ __forceinline__ void wait_async0() {
  asm volatile("s_wait_asynccnt 0x0" ::: "memory");
}

// issue the B-tile copy (TDM on wave 0, or per-lane async fallback)
template <int KDIM>
__device__ __forceinline__ void stage_b(int tid, int wv, unsigned short* bsh,
                                        const unsigned short* __restrict__ Wt, int kk) {
#if TDM_AVAILABLE
  if (wv == 0)
    tdm_load_b_tile((unsigned)(size_t)bsh, Wt + kk, KDIM);
#else
#pragma unroll
  for (int s = tid; s < 1024; s += 256) {
    int n = s >> 2, c = s & 3;
    unsigned lds = (unsigned)(size_t)(bsh + n * LDS_BSTR) + c * 16;
    unsigned long long ga = (unsigned long long)(size_t)(Wt + n * KDIM + kk) + c * 16;
    async_copy16(lds, ga);
  }
#endif
}

// wait for the B-tile copy issued by this wave
__device__ __forceinline__ void stage_b_wait(int wv) {
#if TDM_AVAILABLE
  if (wv == 0) __builtin_amdgcn_s_wait_tensorcnt((short)0);
#else
  wait_async0();
#endif
}

// ---------------- weight convert + transpose: W (K x D) f32 -> Wt (D x K) bf16 ----------------
__global__ void wconv_kernel(const float* __restrict__ W,
                             unsigned short* __restrict__ Wt, int K) {
  int t = blockIdx.x * blockDim.x + threadIdx.x;   // t = k*D + n (coalesced read)
  if (t >= K * D) return;
  int k = t / D;
  int n = t - k * D;
  Wt[n * K + k] = f32_to_bf16(W[t]);
}

// ---------------- embedding gather: Sb[n, d] = bf16(E[token[n], d]) ----------------
__global__ void gather_kernel(const int* __restrict__ tok,
                              const float* __restrict__ E,
                              unsigned short* __restrict__ Sb) {
  int t  = blockIdx.x * blockDim.x + threadIdx.x;  // one float4 per thread
  int n  = t >> 6;                                 // D/4 = 64 chunks per row
  int d4 = (t & 63) << 2;
  float4 e = *(const float4*)(E + tok[n] * D + d4);
  ushort4 o;
  o.x = f32_to_bf16(e.x); o.y = f32_to_bf16(e.y);
  o.z = f32_to_bf16(e.z); o.w = f32_to_bf16(e.w);
  *(ushort4*)(Sb + n * D + d4) = o;
}

// ---------------- fused windowed-context GEMM1: T1 = relu(ctx @ Wc1 + bc1) ----------------
// ctx row i, K index kk: slot p = kk>>8, d = kk&255, j = min(S,i+3)-5+p, valid iff j >= max(0,i-2)
__global__ __launch_bounds__(256) void gemm_ctx_kernel(
    const unsigned short* __restrict__ Sb,    // NROWS x D bf16 (gathered static)
    const unsigned short* __restrict__ Wt,    // D x (CW*D) bf16, N-major
    const float* __restrict__ bias,
    unsigned short* __restrict__ outb) {
  constexpr int KDIM = CW * D;                // 1280
  constexpr int NK   = KDIM / 32;
  __shared__ unsigned short Bsh[2][256 * LDS_BSTR];
  __shared__ unsigned short Ash[2][32 * LDS_ASTR];

  const int tid  = threadIdx.x;
  const int lane = tid & 31;
  const int wv   = tid >> 5;
  const int g    = lane >> 4;
  const int ln16 = lane & 15;
  const int rowBase = blockIdx.x * 32;
  const int row0 = rowBase + (wv & 1) * 16;
  const int c0   = (wv >> 1) * 64;

  v8f acc[4] = {v8f{}, v8f{}, v8f{}, v8f{}};

  auto stage = [&](int buf, int kk) {
    const int p  = kk >> 8;                   // context slot (32-wide block never straddles slots)
    const int d0 = kk & 255;
    stage_b<KDIM>(tid, wv, &Bsh[buf][0], Wt, kk);
    // A tile: 32 rows x 32 K, windowed gather with zero-masking (regular load + ds_store)
    if (tid < 128) {
      int row = tid >> 2, c = tid & 3;
      int m = rowBase + row;
      int i = m & (SEQ - 1);
      int e = (i + 3 < SEQ) ? (i + 3) : SEQ;
      int st = (i - 2 > 0) ? (i - 2) : 0;
      int j = e - CW + p;
      Half8 hv;
      hv.q[0] = 0; hv.q[1] = 0;
      if (j >= st) hv.v = *(const v8bf*)(Sb + (m - i + j) * D + d0 + c * 8);
      *(v8bf*)&Ash[buf][row * LDS_ASTR + c * 8] = hv.v;
    }
  };

  stage(0, 0);
  stage_b_wait(wv);
  __syncthreads();

  for (int kb = 0; kb < NK; ++kb) {
    const int cur = kb & 1;
    if (kb + 1 < NK) stage(cur ^ 1, (kb + 1) * 32);

    Frag a;
    const unsigned short* ab = &Ash[cur][((wv & 1) * 16 + ln16) * LDS_ASTR + g * 8];
    a.h[0] = *(const v8bf*)(ab);
    a.h[1] = *(const v8bf*)(ab + 16);
#pragma unroll
    for (int sub = 0; sub < 4; ++sub) {
      Frag b;
      const unsigned short* bb = &Bsh[cur][(c0 + sub * 16 + ln16) * LDS_BSTR + g * 8];
      b.h[0] = *(const v8bf*)(bb);
      b.h[1] = *(const v8bf*)(bb + 16);
      acc[sub] = __builtin_amdgcn_wmma_f32_16x16x32_bf16(
          false, a.v, false, b.v, (short)0, acc[sub], false, false);
    }
    stage_b_wait(wv);
    __syncthreads();
  }

#pragma unroll
  for (int sub = 0; sub < 4; ++sub) {
    const int n  = c0 + sub * 16 + ln16;
    const float bn = bias[n];
#pragma unroll
    for (int r = 0; r < 8; ++r) {
      const int m = row0 + g * 8 + r;
      float vv = acc[sub][r] + bn;
      vv = fmaxf(vv, 0.0f);                   // ReLU
      outb[m * D + n] = f32_to_bf16(vv);
    }
  }
}

// ---------------- generic GEMM: out = act(A @ Wt^T + bias) ----------------
// A0 (NROWS x 256 bf16); for KDIM==512 the upper K half reads A1 (concat [A0|A1]).
// ACT: 0 = none, 2 = tanh.  OUT_F32: final stage writes f32 to d_out.
template <int KDIM, int ACT, bool OUT_F32>
__global__ __launch_bounds__(256) void gemm_kernel(
    const unsigned short* __restrict__ A0,
    const unsigned short* __restrict__ A1,
    const unsigned short* __restrict__ Wt,    // D x KDIM bf16, N-major
    const float* __restrict__ bias,
    unsigned short* __restrict__ outb,
    float* __restrict__ outf) {
  constexpr int NK = KDIM / 32;
  __shared__ unsigned short Bsh[2][256 * LDS_BSTR];
  __shared__ unsigned short Ash[2][32 * LDS_ASTR];

  const int tid  = threadIdx.x;
  const int lane = tid & 31;
  const int wv   = tid >> 5;
  const int g    = lane >> 4;
  const int ln16 = lane & 15;
  const int rowBase = blockIdx.x * 32;
  const int row0 = rowBase + (wv & 1) * 16;
  const int c0   = (wv >> 1) * 64;

  v8f acc[4] = {v8f{}, v8f{}, v8f{}, v8f{}};

  auto stage = [&](int buf, int kk) {
    stage_b<KDIM>(tid, wv, &Bsh[buf][0], Wt, kk);
    // A tile: coalesced 16B chunks (handles concat source select)
    if (tid < 128) {
      int row = tid >> 2, c = tid & 3;
      const unsigned short* src = A0;
      int kloc = kk;
      if (KDIM > 256 && kk >= 256) { src = A1; kloc = kk - 256; }
      *(v8bf*)&Ash[buf][row * LDS_ASTR + c * 8] =
          *(const v8bf*)(src + (rowBase + row) * D + kloc + c * 8);
    }
  };

  stage(0, 0);
  stage_b_wait(wv);
  __syncthreads();

  for (int kb = 0; kb < NK; ++kb) {
    const int cur = kb & 1;
    if (kb + 1 < NK) stage(cur ^ 1, (kb + 1) * 32);

    Frag a;
    const unsigned short* ab = &Ash[cur][((wv & 1) * 16 + ln16) * LDS_ASTR + g * 8];
    a.h[0] = *(const v8bf*)(ab);
    a.h[1] = *(const v8bf*)(ab + 16);
#pragma unroll
    for (int sub = 0; sub < 4; ++sub) {
      Frag b;
      const unsigned short* bb = &Bsh[cur][(c0 + sub * 16 + ln16) * LDS_BSTR + g * 8];
      b.h[0] = *(const v8bf*)(bb);
      b.h[1] = *(const v8bf*)(bb + 16);
      acc[sub] = __builtin_amdgcn_wmma_f32_16x16x32_bf16(
          false, a.v, false, b.v, (short)0, acc[sub], false, false);
    }
    stage_b_wait(wv);
    __syncthreads();
  }

#pragma unroll
  for (int sub = 0; sub < 4; ++sub) {
    const int n  = c0 + sub * 16 + ln16;
    const float bn = bias[n];
#pragma unroll
    for (int r = 0; r < 8; ++r) {
      const int m = row0 + g * 8 + r;
      float vv = acc[sub][r] + bn;
      if (ACT == 2) vv = tanhf(vv);
      if (OUT_F32) outf[m * D + n] = vv;
      else         outb[m * D + n] = f32_to_bf16(vv);
    }
  }
}

// ---------------- launcher ----------------
extern "C" void kernel_launch(void* const* d_in, const int* in_sizes, int n_in,
                              void* d_out, int out_size, void* d_ws, size_t ws_size,
                              hipStream_t stream) {
  const int*   tok = (const int*)  d_in[0];
  const float* E   = (const float*)d_in[1];
  const float* Wc1 = (const float*)d_in[2];
  const float* bc1 = (const float*)d_in[3];
  const float* Wc2 = (const float*)d_in[4];
  const float* bc2 = (const float*)d_in[5];
  const float* Wr1 = (const float*)d_in[6];
  const float* br1 = (const float*)d_in[7];
  const float* Wr2 = (const float*)d_in[8];
  const float* br2 = (const float*)d_in[9];
  float* out = (float*)d_out;

  // workspace layout (bf16 buffers), ~68.3 MB total
  unsigned short* Sb   = (unsigned short*)d_ws;           // static, NROWS x D
  unsigned short* T1b  = Sb   + (size_t)NROWS * D;        // relu(ctx@Wc1+bc1)
  unsigned short* Hb   = T1b  + (size_t)NROWS * D;        // h
  unsigned short* Rb   = Hb   + (size_t)NROWS * D;        // tanh(comb@Wr1+br1)
  unsigned short* WtC1 = Rb   + (size_t)NROWS * D;        // D x 1280
  unsigned short* WtC2 = WtC1 + (size_t)(CW * D) * D;     // D x 256
  unsigned short* WtR1 = WtC2 + (size_t)D * D;            // D x 512
  unsigned short* WtR2 = WtR1 + (size_t)(2 * D) * D;      // D x 256

  // 1) weight downconvert + transpose to N-major bf16
  wconv_kernel<<<(CW * D * D + 255) / 256, 256, 0, stream>>>(Wc1, WtC1, CW * D);
  wconv_kernel<<<(D * D + 255) / 256,     256, 0, stream>>>(Wc2, WtC2, D);
  wconv_kernel<<<(2 * D * D + 255) / 256, 256, 0, stream>>>(Wr1, WtR1, 2 * D);
  wconv_kernel<<<(D * D + 255) / 256,     256, 0, stream>>>(Wr2, WtR2, D);

  // 2) embedding gather -> bf16
  gather_kernel<<<NROWS * (D / 4) / 256, 256, 0, stream>>>(tok, E, Sb);

  // 3) fused windowed-context GEMM1 (+ReLU)
  gemm_ctx_kernel<<<NROWS / 32, 256, 0, stream>>>(Sb, WtC1, bc1, T1b);

  // 4) GEMM2: h = T1 @ Wc2 + bc2
  gemm_kernel<256, 0, false><<<NROWS / 32, 256, 0, stream>>>(T1b, nullptr, WtC2, bc2, Hb, nullptr);

  // 5) GEMM3: r = tanh([static|h] @ Wr1 + br1)
  gemm_kernel<512, 2, false><<<NROWS / 32, 256, 0, stream>>>(Sb, Hb, WtR1, br1, Rb, nullptr);

  // 6) GEMM4: out = r @ Wr2 + br2 (f32)
  gemm_kernel<256, 0, true><<<NROWS / 32, 256, 0, stream>>>(Rb, nullptr, WtR2, br2, nullptr, out);
}
